// LSTMGATModel_51230369907286
// MI455X (gfx1250) — matrix operand order repaired and checked by reading, not compiled, not run on hardware
//
#include <hip/hip_runtime.h>
#include <math.h>

// Problem dims (fixed by the reference)
#define BDIM   32
#define TDIM   64
#define FDIM   32
#define GDIM   32
#define HDIM   256
#define NNODE  20000
#define CDIM   32
#define NG     1024            // 4*H
#define ROWS   (BDIM * TDIM)   // 2048
#define EPSBN  1e-5f

typedef __attribute__((ext_vector_type(16))) _Float16 v16h;
typedef __attribute__((ext_vector_type(8)))  _Float16 h8;
typedef __attribute__((ext_vector_type(8)))  float    v8f;

// ---------------------------------------------------------------------------
// WMMA helpers (CDNA5, wave32). D = A(16x32 f16) * B(32x16 f16) + C(16x16 f32)
// ---------------------------------------------------------------------------
__device__ __forceinline__ v8f wmma_f16(v16h a, v16h b, v8f c) {
  return __builtin_amdgcn_wmma_f32_16x16x32_f16(
      /*neg_a=*/false, a, /*neg_b=*/false, b,
      /*c_mod=*/(short)0, c, /*reuse_a=*/false, /*reuse_b=*/false);
}

// A fragment, 16x32 f16, source row-major (row M, col K), ld = row stride.
// ISA layout: lane<16: row=lane, halves 0..7 = K 0..7, halves 8..15 = K 16..23
//             lane>=16: row=lane-16, K offset +8.
__device__ __forceinline__ v16h frag_a(const _Float16* base, int ld) {
  int lane = threadIdx.x & 31;
  const _Float16* p = base + (lane & 15) * ld + ((lane >> 4) << 3);
  h8 lo = *(const h8*)(p);
  h8 hi = *(const h8*)(p + 16);
  v16h r;
#pragma unroll
  for (int i = 0; i < 8; ++i) { r[i] = lo[i]; r[i + 8] = hi[i]; }
  return r;
}

// B fragment, 32x16 f16 (K x N). Source is W stored row-major (N rows, K cols),
// i.e. B[k][n] = W[n][k]. ISA layout: lane<16: col=lane, halves 0..7 = K 0..7,
// halves 8..15 = K 16..23; lane>=16: K offset +8.
__device__ __forceinline__ v16h frag_b(const _Float16* base, int ld) {
  int lane = threadIdx.x & 31;
  const _Float16* p = base + (lane & 15) * ld + ((lane >> 4) << 3);
  h8 lo = *(const h8*)(p);
  h8 hi = *(const h8*)(p + 16);
  v16h r;
#pragma unroll
  for (int i = 0; i < 8; ++i) { r[i] = lo[i]; r[i + 8] = hi[i]; }
  return r;
}

__device__ __forceinline__ float sigf(float x) { return 1.0f / (1.0f + __expf(-x)); }

// ---------------------------------------------------------------------------
// f32 -> f16 conversion (weights / activations)
// ---------------------------------------------------------------------------
__global__ void k_f2h(const float* __restrict__ s, _Float16* __restrict__ d, long n) {
  long i = (long)blockIdx.x * blockDim.x + threadIdx.x;
  long stride = (long)gridDim.x * blockDim.x;
  for (; i < n; i += stride) d[i] = (_Float16)s[i];
}

// ---------------------------------------------------------------------------
// Per-channel sum / sum-of-squares, deterministic fixed-order block reduction.
// grid.x = C channels, grid.y = head. stat[head][ch] = sum, stat[head][C+ch] = sumsq
// ---------------------------------------------------------------------------
__global__ __launch_bounds__(256) void k_chan_stats(
    const float* __restrict__ Xbase, long strideX,
    float* __restrict__ statBase, long strideS, int rows, int C) {
  const float* X = Xbase + (long)blockIdx.y * strideX;
  float* stat = statBase + (long)blockIdx.y * strideS;
  int ch = blockIdx.x;
  float s1 = 0.f, s2 = 0.f;
  for (int r = threadIdx.x; r < rows; r += blockDim.x) {
    float v = X[(long)r * C + ch];
    s1 += v; s2 += v * v;
  }
  __shared__ float l1[256], l2[256];
  l1[threadIdx.x] = s1; l2[threadIdx.x] = s2;
  __syncthreads();
  for (int o = 128; o > 0; o >>= 1) {
    if ((int)threadIdx.x < o) {
      l1[threadIdx.x] += l1[threadIdx.x + o];
      l2[threadIdx.x] += l2[threadIdx.x + o];
    }
    __syncthreads();
  }
  if (threadIdx.x == 0) { stat[ch] = l1[0]; stat[C + ch] = l2[0]; }
}

// ---------------------------------------------------------------------------
// Gather + BN(gat) + concat -> lstm_in f16 (ROWS x 64). One wave per (b,t).
// ---------------------------------------------------------------------------
__global__ __launch_bounds__(256) void k_gather(
    const float* __restrict__ feats, const float* __restrict__ gat,
    const int* __restrict__ row_ids, const int* __restrict__ node_ids,
    const float* __restrict__ stat, const float* __restrict__ gamma,
    const float* __restrict__ beta, _Float16* __restrict__ X) {
  int wid = blockIdx.x * (blockDim.x >> 5) + (threadIdx.x >> 5);
  if (wid >= ROWS) return;
  int lane = threadIdx.x & 31;
  int target = row_ids[wid];
  int best = 0x7fffffff;
  for (int n = lane; n < NNODE; n += 32) {
    if (node_ids[n] == target && n < best) best = n;  // first match per lane
  }
#pragma unroll
  for (int o = 16; o > 0; o >>= 1) {
    int other = __shfl_xor(best, o, 32);
    best = best < other ? best : other;
  }
  int b = wid / TDIM;
  // feature half (F == 32 == wave width)
  X[(long)wid * 64 + lane] = (_Float16)feats[(long)wid * FDIM + lane];
  float v = 0.f;
  if (best < NNODE) {
    const float invCnt = 1.0f / (float)(BDIM * NNODE);
    float mean = stat[lane] * invCnt;
    float var  = stat[GDIM + lane] * invCnt - mean * mean;
    float x = gat[((long)b * NNODE + best) * GDIM + lane];
    v = (x - mean) * gamma[lane] * rsqrtf(var + EPSBN) + beta[lane];
  }
  X[(long)wid * 64 + 32 + lane] = (_Float16)v;
}

// ---------------------------------------------------------------------------
// xp = X(f16, MxK) @ W(f16, NxK)^T + bias, output f32 (MxN).
// 256 threads = 8 waves/block, one 32x32 output block per wave.
// grid.y = head (stride-addressed).
// ---------------------------------------------------------------------------
__global__ __launch_bounds__(256) void k_gemm_xp(
    const _Float16* __restrict__ Xbase, long strideX,
    const _Float16* __restrict__ Wbase, long strideW,
    const float* __restrict__ Bbase, long strideB,
    float* __restrict__ Obase, long strideO,
    int M, int N, int K) {
  const _Float16* X = Xbase + (long)blockIdx.y * strideX;
  const _Float16* W = Wbase + (long)blockIdx.y * strideW;
  const float* bias = Bbase + (long)blockIdx.y * strideB;
  float* O = Obase + (long)blockIdx.y * strideO;

  int wid = blockIdx.x * (blockDim.x >> 5) + (threadIdx.x >> 5);
  int nb = N >> 5;
  if (wid >= (M >> 5) * nb) return;
  int bm = wid / nb, bn = wid % nb;
  int lane = threadIdx.x & 31;

  v8f acc00 = {}, acc01 = {}, acc10 = {}, acc11 = {};
  const _Float16* Am = X + (long)(bm * 32) * K;
  const _Float16* Wn = W + (long)(bn * 32) * K;
  for (int k = 0; k < K; k += 32) {
    if (k + 32 < K) {
      __builtin_prefetch(Wn + k + 32, 0, 1);              // global_prefetch
      __builtin_prefetch(Wn + (long)16 * K + k + 32, 0, 1);
    }
    v16h a0 = frag_a(Am + k, K);
    v16h a1 = frag_a(Am + (long)16 * K + k, K);
    v16h b0 = frag_b(Wn + k, K);
    v16h b1 = frag_b(Wn + (long)16 * K + k, K);
    acc00 = wmma_f16(a0, b0, acc00);
    acc01 = wmma_f16(a0, b1, acc01);
    acc10 = wmma_f16(a1, b0, acc10);
    acc11 = wmma_f16(a1, b1, acc11);
  }
  // C/D layout: lane holds col = lane&15, rows r + 8*(lane>>4)
  int c = lane & 15, rb = (lane >> 4) * 8;
  int row0 = bm * 32, col0 = bn * 32;
  float bv0 = bias[col0 + c];
  float bv1 = bias[col0 + 16 + c];
#pragma unroll
  for (int r = 0; r < 8; ++r) {
    O[(long)(row0 + r + rb) * N + col0 + c]           = acc00[r] + bv0;
    O[(long)(row0 + r + rb) * N + col0 + 16 + c]      = acc01[r] + bv1;
    O[(long)(row0 + 16 + r + rb) * N + col0 + c]      = acc10[r] + bv0;
    O[(long)(row0 + 16 + r + rb) * N + col0 + 16 + c] = acc11[r] + bv1;
  }
}

// ---------------------------------------------------------------------------
// Persistent LSTM recurrence. One 1024-thread workgroup (= 32 waves) per head.
// Per timestep: gates(32x1024) = h(32x256) @ Whh(1024x256)^T via WMMA,
// gates staged in LDS, pointwise LSTM update, h/c kept in LDS.
// LDS: h f16 16KB + c f32 32KB + gates f32 128KB = 176KB (CDNA5: 320KB/WGP).
// ---------------------------------------------------------------------------
__global__ __launch_bounds__(1024) void k_lstm_rec(
    const float* __restrict__ xpBase, long strideXp,
    const _Float16* __restrict__ WhhBase, long strideW,
    _Float16* __restrict__ outXBase, long strideOX,
    float* __restrict__ outFBase, long strideOF) {
  const float* xp = xpBase + (long)blockIdx.x * strideXp;
  const _Float16* Whh = WhhBase + (long)blockIdx.x * strideW;
  _Float16* outX = outXBase + (long)blockIdx.x * strideOX;
  float* outF = outFBase ? outFBase + (long)blockIdx.x * strideOF : (float*)0;

  extern __shared__ char smem[];
  _Float16* h_lds = (_Float16*)smem;                              // 32*256 f16
  float* c_lds = (float*)(smem + 32 * 256 * 2);                   // 32*256 f32
  float* g_lds = (float*)(smem + 32 * 256 * 2 + 32 * 256 * 4);    // 32*1024 f32

  int tid = threadIdx.x;
  for (int i = tid; i < 32 * 256; i += 1024) {
    h_lds[i] = (_Float16)0.f;
    c_lds[i] = 0.f;
  }
  __syncthreads();

  int w = tid >> 5, lane = tid & 31;
  int n0 = w * 32;  // this wave's 32 gate columns
  int cc = lane & 15, rb = (lane >> 4) * 8;

  for (int t = 0; t < TDIM; ++t) {
    v8f acc00 = {}, acc01 = {}, acc10 = {}, acc11 = {};
#pragma unroll
    for (int kk = 0; kk < 8; ++kk) {
      int k = kk * 32;
      v16h a0 = frag_a(h_lds + k, HDIM);                  // batch rows 0..15 (LDS)
      v16h a1 = frag_a(h_lds + 16 * HDIM + k, HDIM);      // batch rows 16..31
      v16h b0 = frag_b(Whh + (long)n0 * HDIM + k, HDIM);  // L2-resident weights
      v16h b1 = frag_b(Whh + (long)(n0 + 16) * HDIM + k, HDIM);
      acc00 = wmma_f16(a0, b0, acc00);
      acc01 = wmma_f16(a0, b1, acc01);
      acc10 = wmma_f16(a1, b0, acc10);
      acc11 = wmma_f16(a1, b1, acc11);
    }
#pragma unroll
    for (int r = 0; r < 8; ++r) {
      g_lds[(r + rb) * NG + n0 + cc]             = acc00[r];
      g_lds[(r + rb) * NG + n0 + 16 + cc]        = acc01[r];
      g_lds[(16 + r + rb) * NG + n0 + cc]        = acc10[r];
      g_lds[(16 + r + rb) * NG + n0 + 16 + cc]   = acc11[r];
    }
    __syncthreads();
    // pointwise LSTM cell update: 32*256 = 8192 (b,j) pairs over 1024 threads
#pragma unroll
    for (int q = 0; q < 8; ++q) {
      int p = tid + q * 1024;
      int b = p >> 8, j = p & 255;
      long xrow = (long)(b * TDIM + t) * NG;
      float gi = g_lds[b * NG + j]        + xp[xrow + j];
      float gf = g_lds[b * NG + 256 + j]  + xp[xrow + 256 + j];
      float gg = g_lds[b * NG + 512 + j]  + xp[xrow + 512 + j];
      float go = g_lds[b * NG + 768 + j]  + xp[xrow + 768 + j];
      float cv = sigf(gf) * c_lds[b * 256 + j] + sigf(gi) * tanhf(gg);
      float hv = sigf(go) * tanhf(cv);
      c_lds[b * 256 + j] = cv;
      h_lds[b * 256 + j] = (_Float16)hv;
      long orow = (long)(b * TDIM + t) * HDIM + j;
      outX[orow] = (_Float16)hv;
      if (outF) outF[orow] = hv;
    }
    __syncthreads();
  }
}

// ---------------------------------------------------------------------------
// BN apply -> f16 activations for next GEMM
// ---------------------------------------------------------------------------
__global__ void k_bn_apply(const float* __restrict__ Hb, const float* __restrict__ stat,
                           const float* __restrict__ gamma, const float* __restrict__ beta,
                           _Float16* __restrict__ X, int rows, int C) {
  int n = rows * C;
  float invR = 1.0f / (float)rows;
  for (int i = blockIdx.x * blockDim.x + threadIdx.x; i < n; i += gridDim.x * blockDim.x) {
    int ch = i % C;
    float mean = stat[ch] * invR;
    float var  = stat[C + ch] * invR - mean * mean;
    X[i] = (_Float16)((Hb[i] - mean) * gamma[ch] * rsqrtf(var + EPSBN) + beta[ch]);
  }
}

// ---------------------------------------------------------------------------
// Final: BN(head) at last timestep + FC heads (+softmax for act). One block.
// ---------------------------------------------------------------------------
__global__ __launch_bounds__(256) void k_final(
    const float* __restrict__ H2, long strideH,
    const float* __restrict__ stat, long strideS,
    const float* __restrict__ hg, const float* __restrict__ hb,
    const float* __restrict__ actW, const float* __restrict__ actB,
    const float* __restrict__ tW, const float* __restrict__ tB,
    const float* __restrict__ rW, const float* __restrict__ rB,
    float* __restrict__ out) {
  __shared__ float logits[BDIM * CDIM];
  const float invR = 1.0f / (float)ROWS;
  int tid = threadIdx.x;
  // head 0 -> act logits
  for (int p = tid; p < BDIM * CDIM; p += 256) {
    int b = p / CDIM, c = p % CDIM;
    long row = ((long)b * TDIM + (TDIM - 1)) * HDIM;
    float acc = actB[c];
    for (int j = 0; j < HDIM; ++j) {
      float mean = stat[j] * invR;
      float var  = stat[HDIM + j] * invR - mean * mean;
      float v = (H2[row + j] - mean) * hg[j] * rsqrtf(var + EPSBN) + hb[j];
      acc += v * actW[c * HDIM + j];
    }
    logits[p] = acc;
  }
  __syncthreads();
  if (tid < BDIM) {
    int b = tid;
    float mx = -3.4e38f;
    for (int c = 0; c < CDIM; ++c) mx = fmaxf(mx, logits[b * CDIM + c]);
    float s = 0.f;
    for (int c = 0; c < CDIM; ++c) s += __expf(logits[b * CDIM + c] - mx);
    float inv = 1.0f / s;
    for (int c = 0; c < CDIM; ++c) out[b * CDIM + c] = __expf(logits[b * CDIM + c] - mx) * inv;
  } else if (tid < 2 * BDIM) {
    int b = tid - BDIM;
    const float* H = H2 + strideH;
    const float* st = stat + strideS;
    const float* g = hg + HDIM; const float* be = hb + HDIM;
    long row = ((long)b * TDIM + (TDIM - 1)) * HDIM;
    float acc = tB[0];
    for (int j = 0; j < HDIM; ++j) {
      float mean = st[j] * invR;
      float var  = st[HDIM + j] * invR - mean * mean;
      float v = (H[row + j] - mean) * g[j] * rsqrtf(var + EPSBN) + be[j];
      acc += v * tW[j];
    }
    out[BDIM * CDIM + b] = acc;
  } else if (tid < 3 * BDIM) {
    int b = tid - 2 * BDIM;
    const float* H = H2 + 2 * strideH;
    const float* st = stat + 2 * strideS;
    const float* g = hg + 2 * HDIM; const float* be = hb + 2 * HDIM;
    long row = ((long)b * TDIM + (TDIM - 1)) * HDIM;
    float acc = rB[0];
    for (int j = 0; j < HDIM; ++j) {
      float mean = st[j] * invR;
      float var  = st[HDIM + j] * invR - mean * mean;
      float v = (H[row + j] - mean) * g[j] * rsqrtf(var + EPSBN) + be[j];
      acc += v * rW[j];
    }
    out[BDIM * CDIM + BDIM + b] = acc;
  }
}

// ---------------------------------------------------------------------------
// Host orchestration
// ---------------------------------------------------------------------------
extern "C" void kernel_launch(void* const* d_in, const int* in_sizes, int n_in,
                              void* d_out, int out_size, void* d_ws, size_t ws_size,
                              hipStream_t stream) {
  const float* data_feats = (const float*)d_in[0];
  const float* gat_output = (const float*)d_in[1];
  const int*   row_ids    = (const int*)d_in[2];
  const int*   node_ids   = (const int*)d_in[3];
  const float* sWih0 = (const float*)d_in[4];
  const float* sWhh0 = (const float*)d_in[5];
  const float* sb0   = (const float*)d_in[6];
  const float* sWih1 = (const float*)d_in[7];
  const float* sWhh1 = (const float*)d_in[8];
  const float* sb1   = (const float*)d_in[9];
  const float* hWih  = (const float*)d_in[10];
  const float* hWhh  = (const float*)d_in[11];
  const float* hbias = (const float*)d_in[12];
  const float* bn_g_g = (const float*)d_in[13];
  const float* bn_g_b = (const float*)d_in[14];
  const float* bn_s_g = (const float*)d_in[15];
  const float* bn_s_b = (const float*)d_in[16];
  const float* h_bn_g = (const float*)d_in[17];
  const float* h_bn_b = (const float*)d_in[18];
  const float* actW = (const float*)d_in[19];
  const float* actB = (const float*)d_in[20];
  const float* tW = (const float*)d_in[21];
  const float* tB = (const float*)d_in[22];
  const float* rW = (const float*)d_in[23];
  const float* rB = (const float*)d_in[24];
  float* out = (float*)d_out;

  char* ws = (char*)d_ws;
  size_t off = 0;
  auto take = [&](size_t bytes) -> char* {
    off = (off + 255) & ~(size_t)255;
    char* p = ws + off;
    off += bytes;
    return p;
  };
  const long WH = (long)NG * HDIM;  // 262144 elements per (layer) weight matrix

  float* stat_gnn = (float*)take(64 * 4);
  float* stat_sh  = (float*)take(512 * 4);
  float* stat_hd  = (float*)take(3 * 512 * 4);
  _Float16* Wih0h = (_Float16*)take((size_t)NG * 64 * 2);
  _Float16* Whh0h = (_Float16*)take((size_t)WH * 2);
  _Float16* Wih1h = (_Float16*)take((size_t)WH * 2);
  _Float16* Whh1h = (_Float16*)take((size_t)WH * 2);
  _Float16* hWihh = (_Float16*)take((size_t)6 * WH * 2);
  _Float16* hWhhh = (_Float16*)take((size_t)6 * WH * 2);
  _Float16* Xin   = (_Float16*)take((size_t)ROWS * 64 * 2);
  _Float16* Xa    = (_Float16*)take((size_t)ROWS * HDIM * 2);
  _Float16* Xb    = (_Float16*)take((size_t)ROWS * HDIM * 2);
  _Float16* Xh    = (_Float16*)take((size_t)3 * ROWS * HDIM * 2);
  float* H1 = (float*)take((size_t)ROWS * HDIM * 4);
  float* H2 = (float*)take((size_t)3 * ROWS * HDIM * 4);
  float* xp = (float*)take((size_t)3 * ROWS * NG * 4);

  const size_t REC_LDS = 32 * 256 * 2 + 32 * 256 * 4 + 32 * 1024 * 4;  // 176 KB
  const long XPs = (long)ROWS * NG;
  const long XHs = (long)ROWS * HDIM;

  // Weight conversion to f16 (L2-resident afterwards)
  k_f2h<<<256, 256, 0, stream>>>(sWih0, Wih0h, (long)NG * 64);
  k_f2h<<<256, 256, 0, stream>>>(sWhh0, Whh0h, WH);
  k_f2h<<<256, 256, 0, stream>>>(sWih1, Wih1h, WH);
  k_f2h<<<256, 256, 0, stream>>>(sWhh1, Whh1h, WH);
  k_f2h<<<1024, 256, 0, stream>>>(hWih, hWihh, 6 * WH);
  k_f2h<<<1024, 256, 0, stream>>>(hWhh, hWhhh, 6 * WH);

  // BN stats over the 82MB GAT tensor (single streaming pass), then gather
  k_chan_stats<<<dim3(GDIM, 1), 256, 0, stream>>>(gat_output, 0, stat_gnn, 0,
                                                  BDIM * NNODE, GDIM);
  k_gather<<<ROWS / 8, 256, 0, stream>>>(data_feats, gat_output, row_ids, node_ids,
                                         stat_gnn, bn_g_g, bn_g_b, Xin);

  // shared layer 0 (K=64)
  k_gemm_xp<<<dim3(256, 1), 256, 0, stream>>>(Xin, 0, Wih0h, 0, sb0, 0, xp, 0,
                                              ROWS, NG, 64);
  k_lstm_rec<<<1, 1024, REC_LDS, stream>>>(xp, 0, Whh0h, 0, Xa, 0, (float*)0, 0);

  // shared layer 1 (K=256) -> f32 H1 for BN
  k_gemm_xp<<<dim3(256, 1), 256, 0, stream>>>(Xa, 0, Wih1h, 0, sb1, 0, xp, 0,
                                              ROWS, NG, HDIM);
  k_lstm_rec<<<1, 1024, REC_LDS, stream>>>(xp, 0, Whh1h, 0, Xa, 0, H1, 0);
  k_chan_stats<<<dim3(HDIM, 1), 256, 0, stream>>>(H1, 0, stat_sh, 0, ROWS, HDIM);
  k_bn_apply<<<512, 256, 0, stream>>>(H1, stat_sh, bn_s_g, bn_s_b, Xb, ROWS, HDIM);

  // heads layer 1: 3 independent heads run concurrently (grid dim = 3)
  k_gemm_xp<<<dim3(256, 3), 256, 0, stream>>>(Xb, 0, hWihh, 2 * WH, hbias, 2 * NG,
                                              xp, XPs, ROWS, NG, HDIM);
  k_lstm_rec<<<3, 1024, REC_LDS, stream>>>(xp, XPs, hWhhh, 2 * WH, Xh, XHs,
                                           (float*)0, 0);

  // heads layer 2 -> f32 H2 for BN + FC
  k_gemm_xp<<<dim3(256, 3), 256, 0, stream>>>(Xh, XHs, hWihh + WH, 2 * WH,
                                              hbias + NG, 2 * NG, xp, XPs,
                                              ROWS, NG, HDIM);
  k_lstm_rec<<<3, 1024, REC_LDS, stream>>>(xp, XPs, hWhhh + WH, 2 * WH, Xh, XHs,
                                           H2, XHs);
  k_chan_stats<<<dim3(HDIM, 3), 256, 0, stream>>>(H2, XHs, stat_hd, 512, ROWS, HDIM);

  k_final<<<1, 256, 0, stream>>>(H2, XHs, stat_hd, 512, h_bn_g, h_bn_b,
                                 actW, actB, tW, tB, rW, rB, out);
}
